// CrossAttentionBlock_49804440764888
// MI455X (gfx1250) — matrix operand order repaired and checked by reading, not compiled, not run on hardware
//
#include <hip/hip_runtime.h>
#include <hip/hip_bf16.h>

// ---------------------------------------------------------------------------
// Problem constants (match reference)
// ---------------------------------------------------------------------------
#define B_  4
#define S_  2048
#define A_  512
#define H_  2048
#define NH_ 16
#define DH_ 128          // H_/NH_
#define LN_EPS 1e-5f

typedef __bf16 v16bf __attribute__((ext_vector_type(16)));
typedef __bf16 v8bf  __attribute__((ext_vector_type(8)));
typedef float  v8f   __attribute__((ext_vector_type(8)));

static __device__ __forceinline__ v16bf join8(v8bf lo, v8bf hi) {
    v16bf r;
#pragma unroll
    for (int i = 0; i < 8; ++i) { r[i] = lo[i]; r[i + 8] = hi[i]; }
    return r;
}

static __device__ __forceinline__ v8f wmma_bf16(v16bf a, v16bf b, v8f c) {
    // D = A(16x32 bf16) x B(32x16 bf16) + C(16x16 f32)
    return __builtin_amdgcn_wmma_f32_16x16x32_bf16(false, a, false, b,
                                                   (short)0, c, false, false);
}

// Flat shared-memory pointer -> LDS byte offset (addr[31:0] is the LDS offset
// in the shared aperture; AS3 pointers are plain offsets).
static __device__ __forceinline__ unsigned lds_off(const void* p) {
    return (unsigned)(unsigned long long)p;
}

// Async DMA: 16 bytes global -> LDS, tracked by ASYNCcnt (CDNA5).
static __device__ __forceinline__ void async_copy_b128(unsigned lds_byte,
                                                       const void* gaddr) {
    asm volatile("global_load_async_to_lds_b128 %0, %1, off"
                 :: "v"(lds_byte), "v"(gaddr) : "memory");
}
static __device__ __forceinline__ void wait_async_le3() {
    asm volatile("s_wait_asynccnt 0x3" ::: "memory");
}
static __device__ __forceinline__ void wait_async_0() {
    asm volatile("s_wait_asynccnt 0x0" ::: "memory");
}

// ---------------------------------------------------------------------------
// fp32 -> bf16 elementwise convert
// ---------------------------------------------------------------------------
__global__ void k_cvt_bf16(const float* __restrict__ in, __bf16* __restrict__ out, int n) {
    int i = blockIdx.x * blockDim.x + threadIdx.x;
    int stride = gridDim.x * blockDim.x;
    for (; i < n; i += stride) out[i] = (__bf16)in[i];
}

// ---------------------------------------------------------------------------
// fp32 [K,N] -> bf16 [N,K] (transpose) via LDS tile; K=N=H_
// ---------------------------------------------------------------------------
__global__ void k_cvt_transpose(const float* __restrict__ in, __bf16* __restrict__ out) {
    __shared__ float tile[32][33];
    const int bx = blockIdx.x * 32;     // column block (N)
    const int by = blockIdx.y * 32;     // row block (K)
    const int tx = threadIdx.x & 31;
    const int ty = threadIdx.x >> 5;    // 0..7
#pragma unroll
    for (int i = ty; i < 32; i += 8)
        tile[i][tx] = in[(size_t)(by + i) * H_ + (bx + tx)];
    __syncthreads();
#pragma unroll
    for (int i = ty; i < 32; i += 8)
        out[(size_t)(bx + i) * H_ + (by + tx)] = (__bf16)tile[tx][i];
}

// ---------------------------------------------------------------------------
// WMMA GEMM with async-LDS double-buffered pipeline.
//   C[M,N] = A[M,K] @ W^T[N,K] + bias
// Block: 256 threads = 8 waves (2 x 4), block tile 64M x 128N,
// per-wave tile 32x32 (4 WMMA accumulators).
// K-slices staged to LDS with GLOBAL_LOAD_ASYNC_TO_LDS_B128 (ASYNCcnt),
// two stages in flight.
//   mode 0: store bf16 row-major [M,N]
//   mode 1: store bf16 per-batch transposed: out[(b*N + n)*rows_per_batch + a]
//   mode 2: store f32 [M,N] with residual added (O-projection feeding LN)
// Requires M % 64 == 0, N % 128 == 0, K % 32 == 0.
// ---------------------------------------------------------------------------
__global__ void k_gemm(const __bf16* __restrict__ Amat, const __bf16* __restrict__ Wt,
                       const float* __restrict__ bias, const float* __restrict__ resid,
                       __bf16* __restrict__ outb, float* __restrict__ outf,
                       int M, int N, int K, int mode, int rows_per_batch) {
    __shared__ __align__(16) __bf16 As[2][64 * 32];    // [buf][row*32 + k]  4 KB each
    __shared__ __align__(16) __bf16 Bs[2][128 * 32];   // [buf][n*32 + k]    8 KB each
    const unsigned A_STRIDE = 64 * 32 * 2;             // bytes per A buffer
    const unsigned B_STRIDE = 128 * 32 * 2;            // bytes per B buffer

    const int tid   = threadIdx.x;
    const int wave  = tid >> 5;
    const int lane  = tid & 31;
    const int lrow  = lane & 15;
    const int lhalf = lane >> 4;
    const int wm    = wave >> 2;        // 0..1
    const int wn    = wave & 3;         // 0..3

    const int nbm = M >> 6;             // block count along M
    const int bm  = blockIdx.x % nbm;
    const int bn  = blockIdx.x / nbm;
    const int m0  = bm * 64;
    const int n0  = bn * 128;

    // --- staging assignments -------------------------------------------------
    // A tile: 64 rows x 64 B; thread t -> row = t>>2, 16B chunk = t&3
    const int ar  = tid >> 2;
    const int ac  = (tid & 3) * 8;      // element offset in K-slice
    const __bf16* agp = Amat + (size_t)(m0 + ar) * K + ac;
    const unsigned alds = lds_off(&As[0][0]) + (unsigned)(ar * 32 + ac) * 2;
    // B tile: 128 rows x 64 B = 512 chunks; thread t handles chunks t and t+256
    const int bn0c = tid >> 2,         bk0c = (tid & 3) * 8;
    const int bn1c = (tid + 256) >> 2, bk1c = ((tid + 256) & 3) * 8;
    const __bf16* bgp0 = Wt + (size_t)(n0 + bn0c) * K + bk0c;
    const __bf16* bgp1 = Wt + (size_t)(n0 + bn1c) * K + bk1c;
    const unsigned blds0 = lds_off(&Bs[0][0]) + (unsigned)(bn0c * 32 + bk0c) * 2;
    const unsigned blds1 = lds_off(&Bs[0][0]) + (unsigned)(bn1c * 32 + bk1c) * 2;

    auto stage = [&](int buf, int k0) {
        async_copy_b128(alds  + buf * A_STRIDE, agp  + k0);
        async_copy_b128(blds0 + buf * B_STRIDE, bgp0 + k0);
        async_copy_b128(blds1 + buf * B_STRIDE, bgp1 + k0);
    };

    const int nk = K / 32;
    stage(0, 0);
    if (nk > 1) stage(1, 32);

    v8f acc[2][2];
#pragma unroll
    for (int i = 0; i < 2; ++i)
#pragma unroll
        for (int j = 0; j < 2; ++j) acc[i][j] = (v8f){};

    for (int kt = 0; kt < nk; ++kt) {
        if (kt + 1 < nk) wait_async_le3();   // my stage-kt loads retired
        else             wait_async_0();
        __syncthreads();                      // everyone's stage-kt visible
        const int buf = kt & 1;

        v16bf afr[2], bfr[2];
#pragma unroll
        for (int ms = 0; ms < 2; ++ms) {
            const __bf16* aS = &As[buf][(wm * 32 + ms * 16 + lrow) * 32];
            afr[ms] = ((const v16bf*)aS)[lhalf];
        }
#pragma unroll
        for (int ns = 0; ns < 2; ++ns) {
            const __bf16* bS = &Bs[buf][(wn * 32 + ns * 16 + lrow) * 32 + lhalf * 16];
            bfr[ns] = *(const v16bf*)bS;
        }
#pragma unroll
        for (int ms = 0; ms < 2; ++ms)
#pragma unroll
            for (int ns = 0; ns < 2; ++ns)
                acc[ms][ns] = wmma_bf16(afr[ms], bfr[ns], acc[ms][ns]);

        __syncthreads();                      // all reads of buf done
        if (kt + 2 < nk) stage(buf, (kt + 2) * 32);
    }

    // --- epilogue ------------------------------------------------------------
#pragma unroll
    for (int ns = 0; ns < 2; ++ns) {
        const int col = n0 + wn * 32 + ns * 16 + lrow;
        const float bv = bias[col];
#pragma unroll
        for (int ms = 0; ms < 2; ++ms) {
#pragma unroll
            for (int r = 0; r < 8; ++r) {
                const int m = m0 + wm * 32 + ms * 16 + lhalf * 8 + r;
                float v = acc[ms][ns][r] + bv;
                if (mode == 0) {
                    outb[(size_t)m * N + col] = (__bf16)v;
                } else if (mode == 1) {
                    const int bb = m / rows_per_batch;
                    const int aa = m % rows_per_batch;
                    outb[((size_t)bb * N + col) * rows_per_batch + aa] = (__bf16)v;
                } else {
                    const size_t idx = (size_t)m * N + col;
                    outf[idx] = v + resid[idx];
                }
            }
        }
    }
}

// ---------------------------------------------------------------------------
// Flash attention: one wave per (batch, head, 16-query tile).
//   Qb [B,S,H] bf16, Kb [B,A,H] bf16, Vt [B,H,A] bf16 (pre-transposed),
//   mask [B,A] f32, Cb [B,S,H] bf16 output context.
// ---------------------------------------------------------------------------
__global__ void k_attn(const __bf16* __restrict__ Qb, const __bf16* __restrict__ Kb,
                       const __bf16* __restrict__ Vt, const float* __restrict__ mask,
                       __bf16* __restrict__ Cb) {
    __shared__ __align__(32) __bf16 pbuf[8][16 * 32];   // per-wave P tile (1 KB each)

    const int wave  = threadIdx.x >> 5;
    const int lane  = threadIdx.x & 31;
    const int lrow  = lane & 15;
    const int lhalf = lane >> 4;

    const int QT = S_ / 16;
    const int w  = blockIdx.x * 8 + wave;
    const int qt = w % QT;
    const int t  = w / QT;
    const int hh = t % NH_;
    const int b  = t / NH_;

    // Preload Q fragments: 16 rows x 128 d, 4 A-fragments of 16x32
    const __bf16* qrow = Qb + ((size_t)(b * S_ + qt * 16 + lrow)) * H_ + hh * DH_;
    v16bf aq[4];
#pragma unroll
    for (int c = 0; c < 4; ++c) {
        const v16bf* pq = (const v16bf*)(qrow + c * 32);
        aq[c] = pq[lhalf];
    }

    v8f acc[8];
#pragma unroll
    for (int c = 0; c < 8; ++c) acc[c] = (v8f){};
    float mrow[8], lsum[8];
#pragma unroll
    for (int r = 0; r < 8; ++r) { mrow[r] = -1e30f; lsum[r] = 0.f; }

    const float scale = 0.08838834764831845f;   // 1/sqrt(128)

    for (int kb = 0; kb < A_ / 32; ++kb) {
        // ---- scores: Q[16x128] @ K^T[128x32] -> two 16x16 C fragments
        v8f s0 = {}, s1 = {};
        const __bf16* kp0 = Kb + ((size_t)(b * A_ + kb * 32 + lrow)) * H_ + hh * DH_ + lhalf * 16;
        const __bf16* kp1 = kp0 + (size_t)16 * H_;
#pragma unroll
        for (int c = 0; c < 4; ++c) {
            v16bf b0 = *(const v16bf*)(kp0 + c * 32);
            v16bf b1 = *(const v16bf*)(kp1 + c * 32);
            s0 = wmma_bf16(aq[c], b0, s0);
            s1 = wmma_bf16(aq[c], b1, s1);
        }

        const float mk0 = mask[b * A_ + kb * 32 + lrow]      * -10000.f;
        const float mk1 = mask[b * A_ + kb * 32 + 16 + lrow] * -10000.f;

        // ---- online softmax (rows live within one 16-lane half)
#pragma unroll
        for (int r = 0; r < 8; ++r) {
            float v0 = s0[r] * scale + mk0;
            float v1 = s1[r] * scale + mk1;
            float rm = fmaxf(v0, v1);
#pragma unroll
            for (int off = 1; off < 16; off <<= 1)
                rm = fmaxf(rm, __shfl_xor(rm, off, 32));
            const float mnew = fmaxf(mrow[r], rm);
            const float corr = __expf(mrow[r] - mnew);
            const float p0 = __expf(v0 - mnew);
            const float p1 = __expf(v1 - mnew);
            float rs = p0 + p1;
#pragma unroll
            for (int off = 1; off < 16; off <<= 1)
                rs += __shfl_xor(rs, off, 32);
            lsum[r] = lsum[r] * corr + rs;
            mrow[r] = mnew;
#pragma unroll
            for (int c = 0; c < 8; ++c) acc[c][r] *= corr;
            // stash P (C layout -> row-major LDS tile)
            const int prow = lhalf * 8 + r;
            pbuf[wave][prow * 32 + lrow]      = (__bf16)p0;
            pbuf[wave][prow * 32 + 16 + lrow] = (__bf16)p1;
        }

        // ---- re-read P in A-fragment layout (16x32, K = 32 keys)
        const v16bf* pv = (const v16bf*)(&pbuf[wave][lrow * 32]);
        v16bf ap = pv[lhalf];

        // ---- ctx += P[16x32] @ V[32x128], 8 WMMAs over d-chunks of 16
#pragma unroll
        for (int c = 0; c < 8; ++c) {
            const __bf16* vp = Vt + ((size_t)(b * H_ + hh * DH_ + c * 16 + lrow)) * A_
                                  + kb * 32 + lhalf * 16;
            v16bf bv = *(const v16bf*)vp;
            acc[c] = wmma_bf16(ap, bv, acc[c]);
        }
    }

    // ---- normalize and store context (bf16, for O-projection GEMM)
#pragma unroll
    for (int r = 0; r < 8; ++r) {
        const float inv = 1.0f / lsum[r];
        const int row = qt * 16 + lhalf * 8 + r;
        __bf16* op = Cb + ((size_t)(b * S_ + row)) * H_ + hh * DH_ + lrow;
#pragma unroll
        for (int c = 0; c < 8; ++c) op[c * 16] = (__bf16)(acc[c][r] * inv);
    }
}

// ---------------------------------------------------------------------------
// LayerNorm over last dim (H_), one block per row. X already = out + residual.
// ---------------------------------------------------------------------------
__global__ void k_layernorm(const float* __restrict__ X, const float* __restrict__ gamma,
                            const float* __restrict__ beta, float* __restrict__ out) {
    const int row = blockIdx.x;
    const float* x = X + (size_t)row * H_;
    float s = 0.f, ss = 0.f;
    for (int i = threadIdx.x; i < H_; i += 256) {
        const float v = x[i];
        s += v; ss += v * v;
    }
#pragma unroll
    for (int off = 1; off < 32; off <<= 1) {
        s  += __shfl_xor(s,  off, 32);
        ss += __shfl_xor(ss, off, 32);
    }
    __shared__ float red[16];
    const int wave = threadIdx.x >> 5;
    if ((threadIdx.x & 31) == 0) { red[wave] = s; red[8 + wave] = ss; }
    __syncthreads();
    if (threadIdx.x == 0) {
        float ts = 0.f, tss = 0.f;
        for (int i = 0; i < 8; ++i) { ts += red[i]; tss += red[8 + i]; }
        red[0] = ts; red[8] = tss;
    }
    __syncthreads();
    const float mu  = red[0] / H_;
    const float var = red[8] / H_ - mu * mu;
    const float inv = rsqrtf(var + LN_EPS);
    for (int i = threadIdx.x; i < H_; i += 256)
        out[(size_t)row * H_ + i] = (x[i] - mu) * inv * gamma[i] + beta[i];
}

// ---------------------------------------------------------------------------
// Host launcher
// ---------------------------------------------------------------------------
extern "C" void kernel_launch(void* const* d_in, const int* in_sizes, int n_in,
                              void* d_out, int out_size, void* d_ws, size_t ws_size,
                              hipStream_t stream) {
    (void)in_sizes; (void)n_in; (void)out_size; (void)ws_size;

    const float* hidden = (const float*)d_in[0];   // [B,S,H]
    const float* audio  = (const float*)d_in[1];   // [B,A,H]
    const float* mask   = (const float*)d_in[2];   // [B,A]
    const float* Wq = (const float*)d_in[3];  const float* bq = (const float*)d_in[4];
    const float* Wk = (const float*)d_in[5];  const float* bk = (const float*)d_in[6];
    const float* Wv = (const float*)d_in[7];  const float* bv = (const float*)d_in[8];
    const float* Wo = (const float*)d_in[9];  const float* bo = (const float*)d_in[10];
    const float* gamma = (const float*)d_in[11];
    const float* beta  = (const float*)d_in[12];
    float* outp = (float*)d_out;

    // ---- workspace carve-up (all sizes multiples of 256 B)
    char* w = (char*)d_ws;
    auto carve = [&](size_t bytes) -> void* {
        void* p = (void*)w;
        w += (bytes + 255) & ~(size_t)255;
        return p;
    };
    const size_t nBSH = (size_t)B_ * S_ * H_;   // 16.78M
    const size_t nBAH = (size_t)B_ * A_ * H_;   //  4.19M
    const size_t nHH  = (size_t)H_ * H_;        //  4.19M

    __bf16* hb  = (__bf16*)carve(nBSH * 2);     // hidden bf16
    __bf16* ab  = (__bf16*)carve(nBAH * 2);     // audio bf16
    __bf16* WqT = (__bf16*)carve(nHH * 2);
    __bf16* WkT = (__bf16*)carve(nHH * 2);
    __bf16* WvT = (__bf16*)carve(nHH * 2);
    __bf16* WoT = (__bf16*)carve(nHH * 2);
    __bf16* Qb  = (__bf16*)carve(nBSH * 2);     // Q bf16 [B,S,H]
    __bf16* Kb  = (__bf16*)carve(nBAH * 2);     // K bf16 [B,A,H]
    __bf16* Vt  = (__bf16*)carve(nBAH * 2);     // V bf16 transposed [B,H,A]
    __bf16* Cb  = (__bf16*)carve(nBSH * 2);     // ctx bf16 [B,S,H]
    float*  X   = (float*) carve(nBSH * 4);     // ctx@Wo + bo + hidden (f32)

    // ---- 1) converts
    k_cvt_bf16<<<8192, 256, 0, stream>>>(hidden, hb, (int)nBSH);
    k_cvt_bf16<<<4096, 256, 0, stream>>>(audio,  ab, (int)nBAH);
    dim3 tg(H_ / 32, H_ / 32);
    k_cvt_transpose<<<tg, 256, 0, stream>>>(Wq, WqT);
    k_cvt_transpose<<<tg, 256, 0, stream>>>(Wk, WkT);
    k_cvt_transpose<<<tg, 256, 0, stream>>>(Wv, WvT);
    k_cvt_transpose<<<tg, 256, 0, stream>>>(Wo, WoT);

    // ---- 2) projections  (block tile 64M x 128N)
    const int Mq = B_ * S_;      // 8192
    const int Ma = B_ * A_;      // 2048
    const int blkQ = (Mq / 64) * (H_ / 128);   // 2048
    const int blkA = (Ma / 64) * (H_ / 128);   //  512
    k_gemm<<<blkQ, 256, 0, stream>>>(hb, WqT, bq, nullptr, Qb, nullptr, Mq, H_, H_, 0, 0);
    k_gemm<<<blkA, 256, 0, stream>>>(ab, WkT, bk, nullptr, Kb, nullptr, Ma, H_, H_, 0, 0);
    k_gemm<<<blkA, 256, 0, stream>>>(ab, WvT, bv, nullptr, Vt, nullptr, Ma, H_, H_, 1, A_);

    // ---- 3) attention: B*NH*(S/16) waves = 8192 -> 1024 blocks
    k_attn<<<1024, 256, 0, stream>>>(Qb, Kb, Vt, mask, Cb);

    // ---- 4) output projection + residual (f32)
    k_gemm<<<blkQ, 256, 0, stream>>>(Cb, WoT, bo, hidden, nullptr, X, Mq, H_, H_, 2, 0);

    // ---- 5) LayerNorm -> d_out
    k_layernorm<<<Mq, 256, 0, stream>>>(X, gamma, beta, outp);
}
